// Transformer_6158983102988
// MI455X (gfx1250) — matrix operand order repaired and checked by reading, not compiled, not run on hardware
//
#include <hip/hip_runtime.h>
#include <hip/hip_bf16.h>
#include <stdint.h>
#include <stddef.h>

// ---------------------------------------------------------------------------
// ViT-Base fwd on gfx1250 (MI455X). Compute-bound (~3.5 TFLOP) -> all GEMMs
// via v_wmma_f32_16x16x32_bf16 (bf16 operands, f32 accum). Global->LDS tile
// staging via the Tensor Data Mover with DOUBLE-BUFFERED ping-pong: wave 0
// issues stage i+1's tensor_load_to_lds before computing stage i and waits
// s_wait_tensorcnt<=2 (in-order completion), overlapping DMA with WMMA.
// TDM pad fields produce the bank-conflict-free padded LDS layout directly.
// All WMMA fragments are contiguous (weights stored transposed, V stored
// [dh][tok]) so fragment reads are pure ds_load_b128 / global_load_b128.
// ---------------------------------------------------------------------------

typedef __bf16 bf16_t;
typedef __attribute__((ext_vector_type(16))) __bf16    v16bf;
typedef __attribute__((ext_vector_type(8)))  float     v8f;
typedef __attribute__((ext_vector_type(4)))  unsigned  u32x4;
typedef __attribute__((ext_vector_type(8)))  unsigned  u32x8;

#define NLAYER 12
#define DMODEL 768
#define NHEAD  12
#define DHEAD  64
#define DMLP   3072
#define BATCH  32
#define IMGSZ  384
#define PSZ    16
#define GRIDP  24
#define NPATCH 576                  // 24*24
#define NTOK   577                  // NPATCH + 1
#define KPATCH 2304                 // C*P*P
#define NOUT   1000
#define MROWS  (BATCH * NTOK)       // 18464
#define MPAT   (BATCH * NPATCH)    // 18432

// ---------------- scalar helpers ----------------

__device__ __forceinline__ bf16_t f2bf(float f) {
  unsigned u = __float_as_uint(f);
  u += 0x7FFFu + ((u >> 16) & 1u);            // round-to-nearest-even
  unsigned short h = (unsigned short)(u >> 16);
  return __builtin_bit_cast(bf16_t, h);
}

__device__ __forceinline__ float gelu_exact(float x) {
  return 0.5f * x * (1.0f + erff(x * 0.70710678118654752f));
}

__device__ __forceinline__ float wave_sum32(float v) {
#pragma unroll
  for (int m = 1; m < 32; m <<= 1) v += __shfl_xor(v, m, 32);
  return v;
}

// ---------------- WMMA fragment helpers ----------------

__device__ __forceinline__ v8f wmma_bf16(v16bf a, v16bf b, v8f c) {
  // (neg_a, A, neg_b, B, c_mod, C, reuse_a, reuse_b)
  return __builtin_amdgcn_wmma_f32_16x16x32_bf16(false, a, false, b,
                                                 (short)0, c, false, false);
}

__device__ __forceinline__ v16bf frag_from2(uint4 lo, uint4 hi) {
  struct U { uint4 a, b; } u{lo, hi};
  return __builtin_bit_cast(v16bf, u);
}

// A-layout fragment: lane holds K = {h*8..h*8+7, 16+h*8..16+h*8+7}, h=lane/16
// p = row base (row already selected by lane&15). Two b128 loads.
__device__ __forceinline__ v16bf frag_rowA(const bf16_t* p, int half) {
  const bf16_t* q = p + half * 8;
  return frag_from2(*(const uint4*)q, *(const uint4*)(q + 16));
}

// B-layout fragment: lane holds K = {h*8..h*8+7, 16+h*8..16+h*8+7}. Two b128 loads.
__device__ __forceinline__ v16bf frag_rowB(const bf16_t* p, int half) {
  const bf16_t* q = p + half * 8;
  return frag_from2(*(const uint4*)q, *(const uint4*)(q + 16));
}

// ---------------- Tensor Data Mover (2-D tile -> LDS) ----------------
// D# per CDNA5 ISA 8.3/8.4. data_size=2B, pad_enable so the DMA itself
// produces the padded LDS row stride the fragment readers expect.

struct TdmDesc { u32x4 g0; u32x8 g1; };

__device__ __forceinline__ TdmDesc make_tdm2d(
    unsigned lds_addr, unsigned long long gaddr,
    unsigned dim0, unsigned dim1, unsigned stride0,
    unsigned tile0, unsigned tile1,
    unsigned pad_interval_code, unsigned pad_amount_code) {
  TdmDesc d;
  d.g0[0] = 1u;                                   // count=1 (valid), user mode
  d.g0[1] = lds_addr;                             // LDS byte address
  d.g0[2] = (unsigned)gaddr;                      // global_addr[31:0]
  d.g0[3] = ((unsigned)(gaddr >> 32) & 0x01FFFFFFu) | (2u << 30);  // type=2
  d.g1[0] = (1u << 16)                            // data_size = 2 bytes
          | (1u << 20)                            // pad_enable
          | (pad_interval_code << 22)             // DWORDs between pads
          | (pad_amount_code << 25);              // pad DWORDs - 1
  d.g1[1] = (dim0 & 0xFFFFu) << 16;               // tensor_dim0[15:0]
  d.g1[2] = (dim0 >> 16) | ((dim1 & 0xFFFFu) << 16);
  d.g1[3] = (dim1 >> 16) | (tile0 << 16);         // tile_dim0
  d.g1[4] = tile1 & 0xFFFFu;                      // tile_dim1 (tile_dim2=0)
  d.g1[5] = stride0;                              // tensor_dim0_stride[31:0]
  d.g1[6] = 0u;
  d.g1[7] = 0u;
  return d;
}

__device__ __forceinline__ void tdm_load(const TdmDesc& d) {
  asm volatile("tensor_load_to_lds %0, %1" :: "s"(d.g0), "s"(d.g1) : "memory");
}

__device__ __forceinline__ unsigned lds_addr_of(const void* p) {
  return (unsigned)(uintptr_t)p;   // flat LDS addr[31:0] == LDS offset
}

// ---------------- generic WMMA GEMM ----------------
// C[M,N] = A[M,K](bf16,row-major) * BT[N,K](bf16, B transposed)
// [+bias] [+residual] [gelu] -> outf (fp32) and/or outb (bf16).
// K % 32 == 0, N % 64 == 0.

#define GBM 128
#define GBN 64
#define GBK 32
#define APAD 8    // padded row = 40 bf16 = 80 B (16B aligned, conflict-free)

__device__ __forceinline__ void gemm_issue_tiles(
    unsigned ldsA, unsigned ldsB, const bf16_t* A, const bf16_t* BT,
    int bm0, int bn0, int M, int N, int K, int kb) {
  TdmDesc dA = make_tdm2d(
      ldsA, (unsigned long long)(uintptr_t)(A + (size_t)bm0 * K + kb),
      (unsigned)(K - kb), (unsigned)(M - bm0), (unsigned)K,
      GBK, GBM, 3u /*16 DW rows*/, 3u /*4 DW pad*/);
  tdm_load(dA);
  TdmDesc dB = make_tdm2d(
      ldsB, (unsigned long long)(uintptr_t)(BT + (size_t)bn0 * K + kb),
      (unsigned)(K - kb), (unsigned)(N - bn0), (unsigned)K,
      GBK, GBN, 3u, 3u);
  tdm_load(dB);
}

__global__ __launch_bounds__(256) void gemm_bf16_kernel(
    const bf16_t* __restrict__ A, const bf16_t* __restrict__ BT,
    const float* __restrict__ bias, const float* __restrict__ residual,
    float* __restrict__ outf, bf16_t* __restrict__ outb,
    int M, int N, int K, int act_gelu) {
  __shared__ bf16_t As[2][GBM][GBK + APAD];   // ping-pong, filled by TDM
  __shared__ bf16_t BsT[2][GBN][GBK + APAD];

  const int tid  = threadIdx.x;
  const int lane = tid & 31;
  const int wave = tid >> 5;
  const int wm   = (wave >> 1) * 32;   // 4 waves along M
  const int wn   = (wave & 1) * 32;    // 2 waves along N
  const int bm0  = blockIdx.x * GBM;
  const int bn0  = blockIdx.y * GBN;
  const int half = lane >> 4;
  const int l16  = lane & 15;

  v8f acc[2][2];
#pragma unroll
  for (int i = 0; i < 2; ++i)
#pragma unroll
    for (int j = 0; j < 2; ++j)
#pragma unroll
      for (int e = 0; e < 8; ++e) acc[i][j][e] = 0.0f;

  const unsigned ldsA0 = lds_addr_of(&As[0][0][0]);
  const unsigned ldsA1 = lds_addr_of(&As[1][0][0]);
  const unsigned ldsB0 = lds_addr_of(&BsT[0][0][0]);
  const unsigned ldsB1 = lds_addr_of(&BsT[1][0][0]);

  const int nk = K / GBK;
  if (tid < 32)  // wave 0 drives the DMA; TENSORcnt is per-wave
    gemm_issue_tiles(ldsA0, ldsB0, A, BT, bm0, bn0, M, N, K, 0);

  for (int i = 0; i < nk; ++i) {
    const int cur = i & 1;
    if (tid < 32) {
      if (i + 1 < nk) {  // overlap next DMA with this stage's WMMAs
        gemm_issue_tiles(cur ? ldsA0 : ldsA1, cur ? ldsB0 : ldsB1, A, BT,
                         bm0, bn0, M, N, K, (i + 1) * GBK);
        __builtin_amdgcn_s_wait_tensorcnt(2);  // stage i's pair has landed
      } else {
        __builtin_amdgcn_s_wait_tensorcnt(0);
      }
    }
    // L2 prefetch two K-tiles ahead (tile i+1 is already in flight via TDM)
    if (i + 2 < nk) {
      int kb2 = (i + 2) * GBK;
      int pa = bm0 + (tid >> 1);
      if (pa < M) __builtin_prefetch(A + (size_t)pa * K + kb2, 0, 1);
      int pb = bn0 + (tid >> 2);
      if (pb < N) __builtin_prefetch(BT + (size_t)pb * K + kb2, 0, 1);
    }
    __syncthreads();  // (a) stage data visible to all waves

    v16bf af0 = frag_rowA(&As[cur][wm + l16][0], half);
    v16bf af1 = frag_rowA(&As[cur][wm + 16 + l16][0], half);
    v16bf bf0 = frag_rowB(&BsT[cur][wn + l16][0], half);
    v16bf bf1 = frag_rowB(&BsT[cur][wn + 16 + l16][0], half);

    acc[0][0] = wmma_bf16(af0, bf0, acc[0][0]);
    acc[0][1] = wmma_bf16(af0, bf1, acc[0][1]);
    acc[1][0] = wmma_bf16(af1, bf0, acc[1][0]);
    acc[1][1] = wmma_bf16(af1, bf1, acc[1][1]);

    __syncthreads();  // (b) all done reading; buffer may be overwritten
  }

#pragma unroll
  for (int mi = 0; mi < 2; ++mi)
#pragma unroll
    for (int ni = 0; ni < 2; ++ni) {
      int col = bn0 + wn + ni * 16 + l16;
#pragma unroll
      for (int e = 0; e < 8; ++e) {
        int row = bm0 + wm + mi * 16 + half * 8 + e;
        if (row < M) {
          float v = acc[mi][ni][e];
          if (bias)     v += bias[col];
          if (residual) v += residual[(size_t)row * N + col];
          if (act_gelu) v  = gelu_exact(v);
          size_t o = (size_t)row * N + col;
          if (outf) outf[o] = v;
          if (outb) outb[o] = f2bf(v);
        }
      }
    }
}

// ---------------- flash attention (per batch-head, 64-query tiles) ---------
// Q,K : [B*H, NTOK, 64] bf16 (pre-normalized). Vt : [B*H, 64, NTOK] bf16
// (transposed so P@V B-fragments are contiguous). Out: vh [MROWS, 768].

#define KVPAD 8

__device__ __forceinline__ void attn_issue_tiles(
    unsigned ldsK, unsigned ldsV, const bf16_t* Kh, const bf16_t* Vh,
    int kb0) {
  // K tile: 32 keys x 64 dh; keys beyond NTOK zero-filled by TDM
  TdmDesc dK = make_tdm2d(
      ldsK, (unsigned long long)(uintptr_t)(Kh + (size_t)kb0 * DHEAD),
      DHEAD, (unsigned)(NTOK - kb0), DHEAD,
      DHEAD, 32, 4u /*32 DW rows*/, 3u /*4 DW pad*/);
  tdm_load(dK);
  // V tile: 64 dh-rows x 32 keys (contiguous along tok)
  TdmDesc dV = make_tdm2d(
      ldsV, (unsigned long long)(uintptr_t)(Vh + kb0),
      (unsigned)(NTOK - kb0), DHEAD, NTOK,
      32, DHEAD, 3u /*16 DW rows*/, 3u);
  tdm_load(dV);
}

__global__ __launch_bounds__(128) void attn_kernel(
    const bf16_t* __restrict__ Q, const bf16_t* __restrict__ Kmat,
    const bf16_t* __restrict__ Vt, bf16_t* __restrict__ Ovh) {
  __shared__ bf16_t Ks[2][32][DHEAD + KVPAD];   // [key][dh] ping-pong
  __shared__ bf16_t Vst[2][DHEAD][32 + KVPAD];  // [dh][key] ping-pong
  __shared__ bf16_t Ps[4][16][32 + APAD];

  const int bh = blockIdx.x;
  const int qt = blockIdx.y;
  const int b  = bh / NHEAD;
  const int h  = bh % NHEAD;
  const bf16_t* Qh = Q    + (size_t)bh * NTOK * DHEAD;
  const bf16_t* Kh = Kmat + (size_t)bh * NTOK * DHEAD;
  const bf16_t* Vh = Vt   + (size_t)bh * DHEAD * NTOK;

  const int tid  = threadIdx.x;
  const int lane = tid & 31;
  const int w    = tid >> 5;          // 4 waves, 16 query rows each
  const int half = lane >> 4;
  const int l16  = lane & 15;
  const int q0   = qt * 64 + w * 16;

  // Q A-fragments: two K-chunks of 32 covering DHEAD=64 (global b128s)
  v16bf qf[2];
  {
    int r = q0 + l16;
    if (r >= NTOK) r = NTOK - 1;      // clamp; rows >= NTOK never stored
    const bf16_t* p = Qh + (size_t)r * DHEAD;
#pragma unroll
    for (int ks = 0; ks < 2; ++ks) {
      const bf16_t* qp = p + ks * 32 + half * 8;
      qf[ks] = frag_from2(*(const uint4*)qp, *(const uint4*)(qp + 16));
    }
  }

  float mrow[8], lrow[8];
#pragma unroll
  for (int e = 0; e < 8; ++e) { mrow[e] = -1e30f; lrow[e] = 0.0f; }
  v8f o[4];
#pragma unroll
  for (int t = 0; t < 4; ++t)
#pragma unroll
    for (int e = 0; e < 8; ++e) o[t][e] = 0.0f;

  const unsigned ldsK0 = lds_addr_of(&Ks[0][0][0]);
  const unsigned ldsK1 = lds_addr_of(&Ks[1][0][0]);
  const unsigned ldsV0 = lds_addr_of(&Vst[0][0][0]);
  const unsigned ldsV1 = lds_addr_of(&Vst[1][0][0]);

  const int nkb = (NTOK + 31) / 32;   // 19 key blocks
  if (tid < 32) attn_issue_tiles(ldsK0, ldsV0, Kh, Vh, 0);

  for (int kb = 0; kb < nkb; ++kb) {
    const int kb0 = kb * 32;
    const int cur = kb & 1;
    if (tid < 32) {
      if (kb + 1 < nkb) {  // overlap next KV DMA with this block's compute
        attn_issue_tiles(cur ? ldsK0 : ldsK1, cur ? ldsV0 : ldsV1, Kh, Vh,
                         kb0 + 32);
        __builtin_amdgcn_s_wait_tensorcnt(2);
      } else {
        __builtin_amdgcn_s_wait_tensorcnt(0);
      }
    }
    if (kb0 + 64 < NTOK) {  // prefetch two KV tiles ahead into L2
      __builtin_prefetch(Kh + (size_t)(kb0 + 64 + (tid & 31)) * DHEAD, 0, 1);
      __builtin_prefetch(Vh + (size_t)(tid & 63) * NTOK + kb0 + 64, 0, 1);
    }
    __syncthreads();

    // S = Q K^T  (16 queries x 32 keys)
    v8f s[2];
#pragma unroll
    for (int nt = 0; nt < 2; ++nt)
#pragma unroll
      for (int e = 0; e < 8; ++e) s[nt][e] = 0.0f;
#pragma unroll
    for (int nt = 0; nt < 2; ++nt) {
#pragma unroll
      for (int ks = 0; ks < 2; ++ks) {
        const bf16_t* kp = &Ks[cur][nt * 16 + l16][0] + ks * 32 + half * 8;
        v16bf kf = frag_from2(*(const uint4*)kp, *(const uint4*)(kp + 16));
        s[nt] = wmma_bf16(qf[ks], kf, s[nt]);
      }
    }

    // scale + mask
#pragma unroll
    for (int nt = 0; nt < 2; ++nt) {
      int key = kb0 + nt * 16 + l16;
#pragma unroll
      for (int e = 0; e < 8; ++e) {
        float v = s[nt][e] * 0.125f;  // 1/sqrt(64)
        if (key >= NTOK) v = -1e30f;
        s[nt][e] = v;
      }
    }

    // online softmax (row reductions within each 16-lane half)
    float rm[8];
#pragma unroll
    for (int e = 0; e < 8; ++e) rm[e] = fmaxf(s[0][e], s[1][e]);
#pragma unroll
    for (int m = 1; m < 16; m <<= 1)
#pragma unroll
      for (int e = 0; e < 8; ++e) rm[e] = fmaxf(rm[e], __shfl_xor(rm[e], m, 32));

    float alpha[8], rs[8];
#pragma unroll
    for (int e = 0; e < 8; ++e) {
      float nm = fmaxf(mrow[e], rm[e]);
      alpha[e] = __expf(mrow[e] - nm);
      mrow[e]  = nm;
      rs[e]    = 0.0f;
    }
#pragma unroll
    for (int nt = 0; nt < 2; ++nt)
#pragma unroll
      for (int e = 0; e < 8; ++e) {
        float p = __expf(s[nt][e] - mrow[e]);
        s[nt][e] = p;
        rs[e] += p;
      }
#pragma unroll
    for (int m = 1; m < 16; m <<= 1)
#pragma unroll
      for (int e = 0; e < 8; ++e) rs[e] += __shfl_xor(rs[e], m, 32);
#pragma unroll
    for (int e = 0; e < 8; ++e) lrow[e] = lrow[e] * alpha[e] + rs[e];
#pragma unroll
    for (int t = 0; t < 4; ++t)
#pragma unroll
      for (int e = 0; e < 8; ++e) o[t][e] *= alpha[e];

    // P -> LDS (wave-local), then O += P @ V
#pragma unroll
    for (int nt = 0; nt < 2; ++nt)
#pragma unroll
      for (int e = 0; e < 8; ++e)
        Ps[w][half * 8 + e][nt * 16 + l16] = f2bf(s[nt][e]);

    const bf16_t* pp = &Ps[w][l16][0] + half * 8;
    v16bf pf = frag_from2(*(const uint4*)pp, *(const uint4*)(pp + 16));
#pragma unroll
    for (int nt = 0; nt < 4; ++nt) {
      const bf16_t* vp = &Vst[cur][nt * 16 + l16][0] + half * 8;
      v16bf vf = frag_from2(*(const uint4*)vp, *(const uint4*)(vp + 16));
      o[nt] = wmma_bf16(pf, vf, o[nt]);
    }
    __syncthreads();
  }

  // normalize and scatter to vh[row, h*64 + d]
#pragma unroll
  for (int nt = 0; nt < 4; ++nt)
#pragma unroll
    for (int e = 0; e < 8; ++e) {
      int tok = q0 + half * 8 + e;
      if (tok < NTOK) {
        float v = o[nt][e] / lrow[e];
        Ovh[((size_t)(b * NTOK + tok)) * DMODEL + h * DHEAD + nt * 16 + l16] =
            f2bf(v);
      }
    }
}

// ---------------- LayerNorm: one wave per row (768 = 24 * 32) --------------

__global__ __launch_bounds__(256) void layernorm_kernel(
    const float* __restrict__ X, const float* __restrict__ gam,
    const float* __restrict__ bet, float* __restrict__ Yf,
    bf16_t* __restrict__ Yb, int M) {
  const int lane = threadIdx.x & 31;
  const int w    = threadIdx.x >> 5;
  const int row  = blockIdx.x * 8 + w;
  if (row >= M) return;
  const float* x = X + (size_t)row * DMODEL;
  float v[24];
  float sum = 0.0f;
#pragma unroll
  for (int j = 0; j < 24; ++j) {
    v[j] = x[lane + j * 32];
    sum += v[j];
  }
  float mean = wave_sum32(sum) * (1.0f / DMODEL);
  float var = 0.0f;
#pragma unroll
  for (int j = 0; j < 24; ++j) {
    float d = v[j] - mean;
    var += d * d;
  }
  var = wave_sum32(var) * (1.0f / DMODEL);
  float rstd = rsqrtf(var + 1e-5f);
#pragma unroll
  for (int j = 0; j < 24; ++j) {
    int c = lane + j * 32;
    float y = (v[j] - mean) * rstd * gam[c] + bet[c];
    Yf[(size_t)row * DMODEL + c] = y;
    Yb[(size_t)row * DMODEL + c] = f2bf(y);
  }
}

// ------- QKV prep: cosine-norm q,k per head; scatter ------
// Qb,Kb: [B*H, NTOK, 64].  Vt: [B*H, 64, NTOK] (transposed).

__global__ __launch_bounds__(256) void qkv_prep_kernel(
    const float* __restrict__ QKV, bf16_t* __restrict__ Qb,
    bf16_t* __restrict__ Kb, bf16_t* __restrict__ Vt, int M) {
  const int lane = threadIdx.x & 31;
  const int w    = threadIdx.x >> 5;
  const int row  = blockIdx.x * 8 + w;
  if (row >= M) return;
  const int b   = row / NTOK;
  const int tok = row % NTOK;
  const float* src = QKV + (size_t)row * (3 * DMODEL);
#pragma unroll 1
  for (int h = 0; h < NHEAD; ++h) {
    size_t bhNT = (size_t)(b * NHEAD + h);
    size_t dst  = (bhNT * NTOK + tok) * DHEAD;
    int d0 = h * DHEAD + lane * 2;
    // q
    float q0 = src[d0], q1 = src[d0 + 1];
    float qn = wave_sum32(q0 * q0 + q1 * q1);
    float qi = 1.0f / (sqrtf(qn) + 1e-6f);
    Qb[dst + lane * 2]     = f2bf(q0 * qi);
    Qb[dst + lane * 2 + 1] = f2bf(q1 * qi);
    // k
    float k0 = src[DMODEL + d0], k1 = src[DMODEL + d0 + 1];
    float kn = wave_sum32(k0 * k0 + k1 * k1);
    float ki = 1.0f / (sqrtf(kn) + 1e-6f);
    Kb[dst + lane * 2]     = f2bf(k0 * ki);
    Kb[dst + lane * 2 + 1] = f2bf(k1 * ki);
    // v -> transposed [dh][tok]
    size_t vbase = bhNT * DHEAD * NTOK + tok;
    Vt[vbase + (size_t)(lane * 2) * NTOK]     = f2bf(src[2 * DMODEL + d0]);
    Vt[vbase + (size_t)(lane * 2 + 1) * NTOK] = f2bf(src[2 * DMODEL + d0 + 1]);
  }
}

// ---------------- elementwise kernels ----------------

// dst[l][n][k] = src[l][k][n]  (cast + transpose, weights become B^T)
__global__ void cast_transpose_kernel(const float* __restrict__ src,
                                      bf16_t* __restrict__ dst, int L, int K,
                                      int N) {
  size_t total = (size_t)L * K * N;
  size_t i = (size_t)blockIdx.x * blockDim.x + threadIdx.x;
  if (i >= total) return;
  size_t r = i % ((size_t)K * N);
  size_t l = i / ((size_t)K * N);
  int n = (int)(r / K);
  int k = (int)(r % K);
  dst[i] = f2bf(src[(l * K + k) * (size_t)N + n]);
}

// fused transposed QKV weight: dst[l][n(0..2303)][k(0..767)]
__global__ void pack_qkv_t_kernel(const float* __restrict__ wq,
                                  const float* __restrict__ wk,
                                  const float* __restrict__ wv,
                                  bf16_t* __restrict__ dst) {
  size_t total = (size_t)NLAYER * 3 * DMODEL * DMODEL;
  size_t i = (size_t)blockIdx.x * blockDim.x + threadIdx.x;
  if (i >= total) return;
  size_t r = i % ((size_t)3 * DMODEL * DMODEL);
  size_t l = i / ((size_t)3 * DMODEL * DMODEL);
  int n = (int)(r / DMODEL);
  int k = (int)(r % DMODEL);
  int sel = (n >= 2 * DMODEL) ? 2 : (n >= DMODEL ? 1 : 0);
  int cc  = n - sel * DMODEL;
  const float* wsrc = (sel == 0) ? wq : (sel == 1) ? wk : wv;
  dst[i] = f2bf(wsrc[(l * DMODEL + k) * (size_t)DMODEL + cc]);
}

__global__ void im2col_kernel(const float* __restrict__ img,
                              bf16_t* __restrict__ Apat) {
  size_t total = (size_t)MPAT * KPATCH;
  size_t idx = (size_t)blockIdx.x * blockDim.x + threadIdx.x;
  if (idx >= total) return;
  int col = (int)(idx % KPATCH);
  int row = (int)(idx / KPATCH);
  int b  = row / NPATCH;
  int p  = row % NPATCH;
  int gy = p / GRIDP, gx = p % GRIDP;
  int c  = col / (PSZ * PSZ);
  int r2 = col % (PSZ * PSZ);
  int py = r2 / PSZ, px = r2 % PSZ;
  size_t src = (((size_t)b * 3 + c) * IMGSZ + gy * PSZ + py) * IMGSZ +
               gx * PSZ + px;
  Apat[idx] = f2bf(img[src]);
}

__global__ void assemble_x_kernel(const float* __restrict__ tok,
                                  const float* __restrict__ cls,
                                  const float* __restrict__ pos,
                                  float* __restrict__ X) {
  size_t total = (size_t)MROWS * DMODEL;
  size_t idx = (size_t)blockIdx.x * blockDim.x + threadIdx.x;
  if (idx >= total) return;
  int d = (int)(idx % DMODEL);
  int r = (int)(idx / DMODEL);
  int b = r / NTOK, t = r % NTOK;
  float v;
  if (t == 0)
    v = cls[d] + pos[d];
  else
    v = tok[((size_t)b * NPATCH + (t - 1)) * DMODEL + d] +
        pos[(size_t)t * DMODEL + d];
  X[idx] = v;
}

__global__ void classifier_kernel(const float* __restrict__ X,
                                  const float* __restrict__ wc,
                                  const float* __restrict__ bc,
                                  float* __restrict__ out) {
  int idx = blockIdx.x * blockDim.x + threadIdx.x;
  if (idx >= BATCH * NOUT) return;
  int b = idx / NOUT, oc = idx % NOUT;
  const float* x = X + (size_t)b * NTOK * DMODEL;  // cls token row
  float s = bc[oc];
  for (int d = 0; d < DMODEL; ++d) s += x[d] * wc[(size_t)d * NOUT + oc];
  out[idx] = s;
}

// ---------------- host launcher ----------------

extern "C" void kernel_launch(void* const* d_in, const int* in_sizes, int n_in,
                              void* d_out, int out_size, void* d_ws,
                              size_t ws_size, hipStream_t stream) {
  (void)in_sizes; (void)n_in; (void)out_size;
  const float* image  = (const float*)d_in[0];
  const float* conv_w = (const float*)d_in[1];
  const float* conv_b = (const float*)d_in[2];
  const float* pos    = (const float*)d_in[3];
  const float* cls    = (const float*)d_in[4];
  const float* ln1_s  = (const float*)d_in[5];
  const float* ln1_b  = (const float*)d_in[6];
  const float* wq     = (const float*)d_in[7];
  const float* wk     = (const float*)d_in[8];
  const float* wv     = (const float*)d_in[9];
  const float* wo     = (const float*)d_in[10];
  const float* bo     = (const float*)d_in[11];
  const float* ln2_s  = (const float*)d_in[12];
  const float* ln2_b  = (const float*)d_in[13];
  const float* w1     = (const float*)d_in[14];
  const float* b1     = (const float*)d_in[15];
  const float* w2     = (const float*)d_in[16];
  const float* b2     = (const float*)d_in[17];
  const float* wc     = (const float*)d_in[18];
  const float* bc     = (const float*)d_in[19];
  float* out = (float*)d_out;

  char* ws = (char*)d_ws;
  size_t off = 0;
  auto alloc = [&](size_t bytes) -> char* {
    off = (off + 255) & ~(size_t)255;
    char* p = ws + off;
    off += bytes;
    return p;
  };
  // all weights stored TRANSPOSED: [N][K] bf16
  bf16_t* convWt = (bf16_t*)alloc((size_t)DMODEL * KPATCH * 2);
  bf16_t* qkvWt  = (bf16_t*)alloc((size_t)NLAYER * 3 * DMODEL * DMODEL * 2);
  bf16_t* woWt   = (bf16_t*)alloc((size_t)NLAYER * DMODEL * DMODEL * 2);
  bf16_t* w1Wt   = (bf16_t*)alloc((size_t)NLAYER * DMLP * DMODEL * 2);
  bf16_t* w2Wt   = (bf16_t*)alloc((size_t)NLAYER * DMODEL * DMLP * 2);
  // bufA: Apat bf16 (85MB) / QKV fp32 (170MB) / MLP-hidden bf16 (113MB)
  char*   bufA   = alloc((size_t)MROWS * 3 * DMODEL * 4);
  float*  bufX   = (float*)alloc((size_t)MROWS * DMODEL * 4);
  float*  bufY   = (float*)alloc((size_t)MROWS * DMODEL * 4);
  bf16_t* bufYb  = (bf16_t*)alloc((size_t)MROWS * DMODEL * 2);
  bf16_t* Qb     = (bf16_t*)alloc((size_t)BATCH * NHEAD * NTOK * DHEAD * 2);
  bf16_t* Kb     = (bf16_t*)alloc((size_t)BATCH * NHEAD * NTOK * DHEAD * 2);
  bf16_t* Vt     = (bf16_t*)alloc((size_t)BATCH * NHEAD * DHEAD * NTOK * 2);
  if (off > ws_size) return;  // workspace too small: nothing we can do

  const int TB = 256;
  auto cdiv = [](long a, long b) { return (int)((a + b - 1) / b); };

  // ---- weights -> transposed bf16 (deterministic, recomputed every call) --
  cast_transpose_kernel<<<cdiv((long)KPATCH * DMODEL, TB), TB, 0, stream>>>(
      conv_w, convWt, 1, KPATCH, DMODEL);
  pack_qkv_t_kernel<<<cdiv((long)NLAYER * 3 * DMODEL * DMODEL, TB), TB, 0,
                      stream>>>(wq, wk, wv, qkvWt);
  cast_transpose_kernel<<<cdiv((long)NLAYER * DMODEL * DMODEL, TB), TB, 0,
                          stream>>>(wo, woWt, NLAYER, DMODEL, DMODEL);
  cast_transpose_kernel<<<cdiv((long)NLAYER * DMODEL * DMLP, TB), TB, 0,
                          stream>>>(w1, w1Wt, NLAYER, DMODEL, DMLP);
  cast_transpose_kernel<<<cdiv((long)NLAYER * DMLP * DMODEL, TB), TB, 0,
                          stream>>>(w2, w2Wt, NLAYER, DMLP, DMODEL);

  // ---- patch embedding ----
  im2col_kernel<<<cdiv((long)MPAT * KPATCH, TB), TB, 0, stream>>>(
      image, (bf16_t*)bufA);
  gemm_bf16_kernel<<<dim3(cdiv(MPAT, GBM), DMODEL / GBN), TB, 0, stream>>>(
      (bf16_t*)bufA, convWt, conv_b, nullptr, bufY, nullptr, MPAT, DMODEL,
      KPATCH, 0);
  assemble_x_kernel<<<cdiv((long)MROWS * DMODEL, TB), TB, 0, stream>>>(
      bufY, cls, pos, bufX);

  // ---- transformer blocks ----
  const int lnGrid = cdiv(MROWS, 8);
  for (int l = 0; l < NLAYER; ++l) {
    layernorm_kernel<<<lnGrid, TB, 0, stream>>>(
        bufX, ln1_s + (size_t)l * DMODEL, ln1_b + (size_t)l * DMODEL, bufY,
        bufYb, MROWS);
    gemm_bf16_kernel<<<dim3(cdiv(MROWS, GBM), (3 * DMODEL) / GBN), TB, 0,
                       stream>>>(
        bufYb, qkvWt + (size_t)l * 3 * DMODEL * DMODEL, nullptr, nullptr,
        (float*)bufA, nullptr, MROWS, 3 * DMODEL, DMODEL, 0);
    qkv_prep_kernel<<<lnGrid, TB, 0, stream>>>((float*)bufA, Qb, Kb, Vt,
                                               MROWS);
    attn_kernel<<<dim3(BATCH * NHEAD, cdiv(NTOK, 64)), 128, 0, stream>>>(
        Qb, Kb, Vt, bufYb);
    gemm_bf16_kernel<<<dim3(cdiv(MROWS, GBM), DMODEL / GBN), TB, 0, stream>>>(
        bufYb, woWt + (size_t)l * DMODEL * DMODEL, bo + (size_t)l * DMODEL,
        bufY, bufX, nullptr, MROWS, DMODEL, DMODEL, 0);
    layernorm_kernel<<<lnGrid, TB, 0, stream>>>(
        bufX, ln2_s + (size_t)l * DMODEL, ln2_b + (size_t)l * DMODEL, bufY,
        bufYb, MROWS);
    gemm_bf16_kernel<<<dim3(cdiv(MROWS, GBM), DMLP / GBN), TB, 0, stream>>>(
        bufYb, w1Wt + (size_t)l * DMLP * DMODEL, b1 + (size_t)l * DMLP, nullptr,
        nullptr, (bf16_t*)bufA, MROWS, DMLP, DMODEL, 1);
    gemm_bf16_kernel<<<dim3(cdiv(MROWS, GBM), DMODEL / GBN), TB, 0, stream>>>(
        (bf16_t*)bufA, w2Wt + (size_t)l * DMODEL * DMLP, b2 + (size_t)l * DMODEL,
        bufY, bufX, nullptr, MROWS, DMODEL, DMLP, 0);
  }

  // ---- classifier on cls rows ----
  classifier_kernel<<<cdiv((long)BATCH * NOUT, TB), TB, 0, stream>>>(bufX, wc,
                                                                     bc, out);
}
